// TSPTransformer_25185688224037
// MI455X (gfx1250) — matrix operand order repaired and checked by reading, not compiled, not run on hardware
//
#include <hip/hip_runtime.h>
#include <math.h>

// ---------------- problem constants ----------------
#define BB 128
#define NN 100
#define IN_DIM 8
#define EE 512
#define HH 8
#define DHH 64
#define FFD 2048
#define LL 6
#define MAX_STEPS 150        // int(N*1.5)
#define ROWS (BB*NN)         // 12800
#define LN_EPS 1e-5f

typedef __attribute__((ext_vector_type(16))) _Float16 v16h;
typedef __attribute__((ext_vector_type(8)))  _Float16 v8h;
typedef __attribute__((ext_vector_type(8)))  float    v8f;
typedef int v4i_raw __attribute__((vector_size(16)));   // matches builtin param

#define ASG __attribute__((address_space(1)))
#define ASL __attribute__((address_space(3)))

#if defined(__gfx1250__) && __has_builtin(__builtin_amdgcn_global_load_async_to_lds_b128) && __has_builtin(__builtin_amdgcn_s_wait_asynccnt)
#define GEMM_ASYNC 1
#endif

__device__ __forceinline__ float sigmf(float x) { return 1.f / (1.f + __expf(-x)); }

// =====================================================================
// WMMA GEMM:  C[M,N] = act( Ah[M,K](f16) @ Wt[N,K](f16)^T + bias
//                           + residual (+C if accum) )
// Wt is the PRE-TRANSPOSED weight (row n holds column n of W[K,N]) so
// both A and B tiles are contiguous-K copies -> b128 global->LDS moves
// (async when the builtin is available) and b128 LDS fragment loads.
// One wave = one 16x16 tile via v_wmma_f32_16x16x32_f16.
// Requires M%16==0, N%16==0, K%32==0.
// =====================================================================
__global__ void __launch_bounds__(256)
gemm_wmma(const _Float16* __restrict__ Ah, const _Float16* __restrict__ Wt,
          const float* __restrict__ bias, const float* __restrict__ R,
          float* __restrict__ C, _Float16* __restrict__ Ch,
          int M, int N, int K, int relu, int resMode, int accum)
{
    __shared__ __align__(16) _Float16 sA[8][2][512];   // [wave][buf][16x32]
    __shared__ __align__(16) _Float16 sB[8][2][512];
    const int wave = threadIdx.x >> 5, lane = threadIdx.x & 31;
    const int n0 = blockIdx.x * 16;
    const int m0 = (blockIdx.y * 8 + wave) * 16;
    if (m0 >= M) return;                       // wave-uniform
    const int g = lane >> 4, mr = lane & 15;
    const int r1 = lane >> 2, r2 = r1 + 8, q = (lane & 3) * 8;  // 16B chunk map

    auto stage = [&](int buf, int k0) {
        const _Float16* ga1 = Ah + (size_t)(m0 + r1) * K + k0 + q;
        const _Float16* ga2 = Ah + (size_t)(m0 + r2) * K + k0 + q;
        const _Float16* gb1 = Wt + (size_t)(n0 + r1) * K + k0 + q;
        const _Float16* gb2 = Wt + (size_t)(n0 + r2) * K + k0 + q;
        _Float16* la1 = &sA[wave][buf][r1 * 32 + q];
        _Float16* la2 = &sA[wave][buf][r2 * 32 + q];
        _Float16* lb1 = &sB[wave][buf][r1 * 32 + q];
        _Float16* lb2 = &sB[wave][buf][r2 * 32 + q];
#ifdef GEMM_ASYNC
        __builtin_amdgcn_global_load_async_to_lds_b128((ASG v4i_raw*)ga1, (ASL v4i_raw*)la1, 0, 0);
        __builtin_amdgcn_global_load_async_to_lds_b128((ASG v4i_raw*)ga2, (ASL v4i_raw*)la2, 0, 0);
        __builtin_amdgcn_global_load_async_to_lds_b128((ASG v4i_raw*)gb1, (ASL v4i_raw*)lb1, 0, 0);
        __builtin_amdgcn_global_load_async_to_lds_b128((ASG v4i_raw*)gb2, (ASL v4i_raw*)lb2, 0, 0);
#else
        *(int4*)la1 = *(const int4*)ga1;
        *(int4*)la2 = *(const int4*)ga2;
        *(int4*)lb1 = *(const int4*)gb1;
        *(int4*)lb2 = *(const int4*)gb2;
#endif
    };

    v8f acc = {};
    auto compute = [&](int buf) {
        // A frag (16x32 f16): lane row mr; elems 0-7 K=8g+0..7, 8-15 K=16+8g+0..7
        const _Float16* pa = &sA[wave][buf][mr * 32];
        v8h alo = *(const v8h*)(pa + 8 * g);
        v8h ahi = *(const v8h*)(pa + 16 + 8 * g);
        // B frag (32x16 f16): lane col mr; elems 0-7 K=8g+0..7, 8-15 K=16+8g+0..7 (Wt row = col)
        const _Float16* pb = &sB[wave][buf][mr * 32];
        v8h blo = *(const v8h*)(pb + 8 * g);
        v8h bhi = *(const v8h*)(pb + 16 + 8 * g);
        v16h av = __builtin_shufflevector(alo, ahi, 0,1,2,3,4,5,6,7,8,9,10,11,12,13,14,15);
        v16h bv = __builtin_shufflevector(blo, bhi, 0,1,2,3,4,5,6,7,8,9,10,11,12,13,14,15);
        acc = __builtin_amdgcn_wmma_f32_16x16x32_f16(false, av, false, bv,
                                                     (short)0, acc, false, false);
    };

#ifdef GEMM_ASYNC
    stage(0, 0);
    int buf = 0;
    for (int k0 = 0; k0 < K; k0 += 32) {
        if (k0 + 32 < K) {
            stage(buf ^ 1, k0 + 32);                 // prefetch next chunk
            __builtin_amdgcn_s_wait_asynccnt(4);     // current chunk landed
        } else {
            __builtin_amdgcn_s_wait_asynccnt(0);
        }
        compute(buf);
        buf ^= 1;
    }
#else
    for (int k0 = 0; k0 < K; k0 += 32) { stage(0, k0); compute(0); }
#endif

    const int col = n0 + mr;
    #pragma unroll
    for (int v = 0; v < 8; ++v) {
        int m = m0 + v + 8 * g;                     // C/D layout: VGPR v -> row v+8g
        float val = acc[v] + bias[col];
        if (resMode == 1)      val += R[(size_t)m * N + col];
        else if (resMode == 2) val += R[(size_t)(m % NN) * N + col];
        if (accum)             val += C[(size_t)m * N + col];
        if (relu)              val = fmaxf(val, 0.f);
        if (C)  C[(size_t)m * N + col] = val;
        if (Ch) Ch[(size_t)m * N + col] = (_Float16)val;
    }
}

// =====================================================================
// One-time weight transform: Wt[n][k] = (f16) W[k][n]   (tiled in LDS)
// =====================================================================
__global__ void __launch_bounds__(256)
transpose_cast_k(const float* __restrict__ W, _Float16* __restrict__ Wt, int K, int N)
{
    __shared__ _Float16 t[32][33];
    int n0 = blockIdx.x * 32, k0 = blockIdx.y * 32;
    int tx = threadIdx.x, ty = threadIdx.y;            // 32 x 8
    #pragma unroll
    for (int j = 0; j < 32; j += 8)
        t[ty + j][tx] = (_Float16)W[(size_t)(k0 + ty + j) * N + (n0 + tx)];
    __syncthreads();
    #pragma unroll
    for (int j = 0; j < 32; j += 8)
        Wt[(size_t)(n0 + ty + j) * K + (k0 + tx)] = t[tx][ty + j];
}

// =====================================================================
// Input projection (K=8, fp32) + positional encoding; dual f32/f16 out
// =====================================================================
__global__ void input_proj_k(const float* __restrict__ nf, const float* __restrict__ w,
                             const float* __restrict__ b, const float* __restrict__ pe,
                             float* __restrict__ X, _Float16* __restrict__ Xh)
{
    int idx = blockIdx.x * 256 + threadIdx.x;
    if (idx < ROWS * EE) {
        int m = idx >> 9, n = idx & 511;
        const float* f = nf + (size_t)m * IN_DIM;
        float s = b[n] + pe[(size_t)(m % NN) * EE + n];
        #pragma unroll
        for (int k = 0; k < IN_DIM; ++k) s += f[k] * w[(size_t)k * EE + n];
        X[idx] = s; Xh[idx] = (_Float16)s;
    }
}

// =====================================================================
// Row LayerNorm over E=512 (one wave per row); nullable f32/f16 outputs
// =====================================================================
__global__ void __launch_bounds__(256)
layernorm_k(const float* __restrict__ X, const float* __restrict__ gam,
            const float* __restrict__ bet, float* __restrict__ Y,
            _Float16* __restrict__ Yh, int rows)
{
    int wave = threadIdx.x >> 5, lane = threadIdx.x & 31;
    int row = blockIdx.x * 8 + wave;
    if (row >= rows) return;
    const float* x = X + (size_t)row * EE;
    float v[16]; float s = 0.f;
    #pragma unroll
    for (int i = 0; i < 16; ++i) { v[i] = x[lane + 32 * i]; s += v[i]; }
    #pragma unroll
    for (int off = 16; off > 0; off >>= 1) s += __shfl_xor(s, off, 32);
    float mu = s * (1.f / EE);
    float ss = 0.f;
    #pragma unroll
    for (int i = 0; i < 16; ++i) { float d = v[i] - mu; ss += d * d; }
    #pragma unroll
    for (int off = 16; off > 0; off >>= 1) ss += __shfl_xor(ss, off, 32);
    float inv = rsqrtf(ss * (1.f / EE) + LN_EPS);
    #pragma unroll
    for (int i = 0; i < 16; ++i) {
        int e = lane + 32 * i;
        float o = (v[i] - mu) * inv * gam[e] + bet[e];
        if (Y)  Y[(size_t)row * EE + e] = o;
        if (Yh) Yh[(size_t)row * EE + e] = (_Float16)o;
    }
}

// =====================================================================
// Encoder self-attention, one block per (b,h); fp32; f16 output
// =====================================================================
__global__ void __launch_bounds__(128)
enc_attn_k(const float* __restrict__ Qf, const float* __restrict__ Kf,
           const float* __restrict__ Vf, _Float16* __restrict__ Oh)
{
    __shared__ float Ps[NN * NN];   // 40 KB probs scratch
    int bh = blockIdx.x, b = bh >> 3, h = bh & 7;
    int tid = threadIdx.x;
    if (tid < NN) {
        const float* q = Qf + ((size_t)(b * NN + tid)) * EE + h * DHH;
        float qr[DHH];
        #pragma unroll
        for (int d = 0; d < DHH; ++d) qr[d] = q[d];
        float mx = -1e30f;
        for (int n = 0; n < NN; ++n) {
            const float* kr = Kf + ((size_t)(b * NN + n)) * EE + h * DHH;
            float s = 0.f;
            #pragma unroll 8
            for (int d = 0; d < DHH; ++d) s += qr[d] * kr[d];
            s *= 0.125f;
            Ps[tid * NN + n] = s;
            mx = fmaxf(mx, s);
        }
        float sum = 0.f;
        for (int n = 0; n < NN; ++n) {
            float e = __expf(Ps[tid * NN + n] - mx);
            Ps[tid * NN + n] = e; sum += e;
        }
        float inv = 1.f / sum;
        _Float16* orow = Oh + ((size_t)(b * NN + tid)) * EE + h * DHH;
        for (int d = 0; d < DHH; ++d) {
            float o = 0.f;
            for (int n = 0; n < NN; ++n)
                o += Ps[tid * NN + n] * Vf[((size_t)(b * NN + n)) * EE + h * DHH + d];
            orow[d] = (_Float16)(o * inv);
        }
    }
}

// =====================================================================
// Decoder 1-query attention, one block per (b,h); f16 output
// =====================================================================
__global__ void __launch_bounds__(128)
dec_attn_k(const float* __restrict__ Qv, const float* __restrict__ Kf,
           const float* __restrict__ Vf, _Float16* __restrict__ Oh)
{
    __shared__ float sc[128];
    __shared__ float pr[128];
    int bh = blockIdx.x, b = bh >> 3, h = bh & 7;
    int tid = threadIdx.x;
    const float* q = Qv + (size_t)b * EE + h * DHH;
    float s = -1e30f;
    if (tid < NN) {
        const float* kr = Kf + ((size_t)(b * NN + tid)) * EE + h * DHH;
        float d = 0.f;
        #pragma unroll 8
        for (int e = 0; e < DHH; ++e) d += q[e] * kr[e];
        s = d * 0.125f;
    }
    sc[tid] = s; __syncthreads();
    for (int off = 64; off > 0; off >>= 1) {
        if (tid < off) sc[tid] = fmaxf(sc[tid], sc[tid + off]);
        __syncthreads();
    }
    float mx = sc[0]; __syncthreads();
    float p = (tid < NN) ? __expf(s - mx) : 0.f;
    pr[tid] = p; sc[tid] = p; __syncthreads();
    for (int off = 64; off > 0; off >>= 1) {
        if (tid < off) sc[tid] += sc[tid + off];
        __syncthreads();
    }
    float inv = 1.f / sc[0];
    if (tid < DHH) {
        float o = 0.f;
        for (int n = 0; n < NN; ++n)
            o += pr[n] * Vf[((size_t)(b * NN + n)) * EE + h * DHH + tid];
        Oh[(size_t)b * EE + h * DHH + tid] = (_Float16)(o * inv);
    }
}

// =====================================================================
// LSTM pointwise (dual h out), done-gate, ctx init, zero fills
// =====================================================================
__global__ void lstm_elem_k(const float* __restrict__ gates,
                            float* __restrict__ h, float* __restrict__ c,
                            _Float16* __restrict__ hh)
{
    int idx = blockIdx.x * 256 + threadIdx.x;
    if (idx < BB * EE) {
        int b = idx >> 9, e = idx & 511;
        const float* g = gates + (size_t)b * (4 * EE);
        float ig = g[e], fg = g[EE + e], gg = g[2 * EE + e], og = g[3 * EE + e];
        float cn = sigmf(fg) * c[idx] + sigmf(ig) * tanhf(gg);
        float hv = sigmf(og) * tanhf(cn);
        c[idx] = cn; h[idx] = hv; hh[idx] = (_Float16)hv;
    }
}

__global__ void done_update_k(const float* __restrict__ h,
                              const float* __restrict__ dw, const float* __restrict__ db,
                              const int* __restrict__ ucount, int* __restrict__ complete,
                              int step)
{
    int b = threadIdx.x;
    if (b < BB) {
        const float* hr = h + (size_t)b * EE;
        float d = 0.f;
        for (int e = 0; e < EE; ++e) d += hr[e] * dw[e];
        d += db[0];
        float p = sigmf(d);
        int eligible = (!complete[b]) && (ucount[b] >= NN) && (step >= NN);
        if (eligible && p > 0.5f) complete[b] = 1;
    }
}

__global__ void ctx_init_k(const float* __restrict__ ne, _Float16* __restrict__ ctxh)
{
    int idx = blockIdx.x * 256 + threadIdx.x;
    if (idx < BB * EE) {
        int b = idx >> 9, e = idx & 511;
        float s = 0.f;
        for (int n = 0; n < NN; ++n) s += ne[((size_t)(b * NN + n)) * EE + e];
        ctxh[idx] = (_Float16)(s * (1.f / NN));
    }
}

__global__ void zero_f_k(float* p, int n)
{ int i = blockIdx.x * 256 + threadIdx.x; if (i < n) p[i] = 0.f; }
__global__ void zero_i_k(int* p, int n)
{ int i = blockIdx.x * 256 + threadIdx.x; if (i < n) p[i] = 0; }
__global__ void zero_h_k(_Float16* p, int n)
{ int i = blockIdx.x * 256 + threadIdx.x; if (i < n) p[i] = (_Float16)0.f; }

// =====================================================================
// Pointer scoring + softmax + argmax + tour-state update (1 block / b)
// =====================================================================
__global__ void __launch_bounds__(128)
select_k(const float* __restrict__ logits, const float* __restrict__ node_emb,
         const float* __restrict__ ew,
         int* __restrict__ visited, int* __restrict__ ucount, int* __restrict__ complete,
         int* __restrict__ all_vis, int* __restrict__ first, int* __restrict__ prevn,
         _Float16* __restrict__ ctxh, const float* __restrict__ sc_bonus,
         const float* __restrict__ rev_pen,
         float* __restrict__ logps_out, int* __restrict__ tours_out, int step)
{
    __shared__ float red[128];
    __shared__ float av[128];
    __shared__ int   ai[128];
    int b = blockIdx.x, tid = threadIdx.x;
    int comp = complete[b], avis = all_vis[b], pv = prevn[b], fst = first[b];

    float s = -1e30f;
    if (tid < NN) {
        const float* lg = logits + (size_t)b * EE;
        const float* ne = node_emb + ((size_t)(b * NN + tid)) * EE;
        float d = 0.f;
        #pragma unroll 8
        for (int e = 0; e < EE; ++e) d += lg[e] * ne[e];
        s = d;
        if (!avis && visited[b * NN + tid]) s = rev_pen[0];
        if (avis) {
            float ewp    = ew[((size_t)(b * NN + pv)) * NN + tid];
            float direct = ew[((size_t)(b * NN + pv)) * NN + 0];
            float via    = ewp + ew[((size_t)(b * NN + tid)) * NN + 0];
            if (via < direct) s += sc_bonus[0] * (direct - via) / direct;
        }
        if (comp) s = -1e9f;
        if (comp && tid == fst) s = 100.0f;
    }
    red[tid] = s; __syncthreads();
    for (int off = 64; off > 0; off >>= 1) {
        if (tid < off) red[tid] = fmaxf(red[tid], red[tid + off]);
        __syncthreads();
    }
    float mx = red[0]; __syncthreads();
    float p = (tid < NN) ? __expf(s - mx) : 0.f;
    red[tid] = p; __syncthreads();
    for (int off = 64; off > 0; off >>= 1) {
        if (tid < off) red[tid] += red[tid + off];
        __syncthreads();
    }
    float sum = red[0];
    av[tid] = (tid < NN) ? p : -1.f; ai[tid] = tid; __syncthreads();
    for (int off = 64; off > 0; off >>= 1) {
        if (tid < off) {
            float po = av[tid + off]; int io = ai[tid + off];
            if (po > av[tid] || (po == av[tid] && io < ai[tid])) { av[tid] = po; ai[tid] = io; }
        }
        __syncthreads();
    }
    int curr = ai[0];
    if (tid == 0) {
        float prob = av[0] / sum;
        logps_out[step * BB + b] = logf(prob + 1e-10f);
        tours_out[step * BB + b] = curr;
        if (step == 0) fst = curr;
        int active = !comp;
        int newly = active && !visited[b * NN + curr];
        int uc = ucount[b] + (newly ? 1 : 0);
        ucount[b] = uc;
        if (active) visited[b * NN + curr] = 1;
        if (uc >= NN) all_vis[b] = 1;
        first[b] = fst;
        prevn[b] = curr;
    }
    __syncthreads();
    int cur = ai[0];
    for (int e = tid; e < EE; e += 128)
        ctxh[(size_t)b * EE + e] = (_Float16)node_emb[((size_t)(b * NN + cur)) * EE + e];
}

__global__ void finalize_k(const int* __restrict__ complete, const int* __restrict__ first,
                           float* __restrict__ logps_out, int* __restrict__ tours_out)
{
    int b = threadIdx.x;
    if (b < BB) {
        tours_out[MAX_STEPS * BB + b] = complete[b] ? 0 : first[b];
        logps_out[MAX_STEPS * BB + b] = 0.f;
    }
}

// =====================================================================
// host-side orchestration
// =====================================================================
static inline void gemm(hipStream_t st, const _Float16* Ah, const _Float16* Wt,
                        const float* bias, const float* R, float* C, _Float16* Ch,
                        int M, int N, int K, int relu, int resMode, int accum)
{
    dim3 grid(N / 16, (M + 127) / 128);
    gemm_wmma<<<grid, 256, 0, st>>>(Ah, Wt, bias, R, C, Ch, M, N, K, relu, resMode, accum);
}

static inline void xpose(hipStream_t st, const float* W, _Float16* Wt, int K, int N)
{
    dim3 grid(N / 32, K / 32), blk(32, 8);
    transpose_cast_k<<<grid, blk, 0, st>>>(W, Wt, K, N);
}

extern "C" void kernel_launch(void* const* d_in, const int* in_sizes, int n_in,
                              void* d_out, int out_size, void* d_ws, size_t ws_size,
                              hipStream_t stream)
{
    (void)in_sizes; (void)n_in; (void)out_size; (void)ws_size;
    const float* node_features = (const float*)d_in[0];
    const float* edge_weights  = (const float*)d_in[1];
    const float* pe            = (const float*)d_in[2];
    const float* in_w          = (const float*)d_in[3];
    const float* in_b          = (const float*)d_in[4];
    const float* enc_wq = (const float*)d_in[5];
    const float* enc_wk = (const float*)d_in[6];
    const float* enc_wv = (const float*)d_in[7];
    const float* enc_wo = (const float*)d_in[8];
    const float* enc_bq = (const float*)d_in[9];
    const float* enc_bk = (const float*)d_in[10];
    const float* enc_bv = (const float*)d_in[11];
    const float* enc_bo = (const float*)d_in[12];
    const float* enc_w1 = (const float*)d_in[13];
    const float* enc_b1 = (const float*)d_in[14];
    const float* enc_w2 = (const float*)d_in[15];
    const float* enc_b2 = (const float*)d_in[16];
    const float* ln1g   = (const float*)d_in[17];
    const float* ln1b   = (const float*)d_in[18];
    const float* ln2g   = (const float*)d_in[19];
    const float* ln2b   = (const float*)d_in[20];
    const float* lstm_wih = (const float*)d_in[21];
    const float* lstm_whh = (const float*)d_in[22];
    const float* lstm_bih = (const float*)d_in[23];
    const float* lstm_bhh = (const float*)d_in[24];
    const float* dec_wq = (const float*)d_in[25];
    const float* dec_wk = (const float*)d_in[26];
    const float* dec_wv = (const float*)d_in[27];
    const float* dec_wo = (const float*)d_in[28];
    const float* dec_pw = (const float*)d_in[29];
    const float* dec_bq = (const float*)d_in[30];
    const float* dec_bk = (const float*)d_in[31];
    const float* dec_bv = (const float*)d_in[32];
    const float* dec_bo = (const float*)d_in[33];
    const float* dec_pb = (const float*)d_in[34];
    const float* dec_lng = (const float*)d_in[35];
    const float* dec_lnb = (const float*)d_in[36];
    const float* shortcut_bonus  = (const float*)d_in[37];
    const float* revisit_penalty = (const float*)d_in[38];
    const float* done_w = (const float*)d_in[39];
    const float* done_b = (const float*)d_in[40];

    float* logps_f = (float*)d_out;                       // (151,128) fp32
    int*   tours_i = (int*)d_out + (MAX_STEPS + 1) * BB;  // (151,128) int32

    // ---- workspace carve-up ----
    const size_t S = (size_t)ROWS * EE;
    float* fp = (float*)d_ws;
    float* xbuf = fp; fp += S;                 // node embeddings / running x (f32)
    float* qb   = fp; fp += S;                 // Q f32 (aliased as residual temp tb)
    float* kb   = fp; fp += S;                 // K f32 (later decoder K)
    float* vb   = fp; fp += S;                 // V f32 (later decoder V)
    float* gatesb = fp; fp += (size_t)BB * 4 * EE;
    float* hb   = fp; fp += (size_t)BB * EE;
    float* cb   = fp; fp += (size_t)BB * EE;
    float* qdec = fp; fp += (size_t)BB * EE;
    float* odec = fp; fp += (size_t)BB * EE;
    float* lgt  = fp; fp += (size_t)BB * EE;
    float* tb   = qb;

    _Float16* hp = (_Float16*)fp;
    _Float16* xh  = hp; hp += S;                // x (f16)
    _Float16* abh = hp; hp += S;                // encoder attn out (f16)
    _Float16* ffh = hp; hp += (size_t)ROWS * FFD;
    _Float16* hh   = hp; hp += (size_t)BB * EE;
    _Float16* ctxh = hp; hp += (size_t)BB * EE;
    _Float16* adh  = hp; hp += (size_t)BB * EE;
    _Float16* o2h  = hp; hp += (size_t)BB * EE;
    // pre-transposed f16 weights
    _Float16 *wqT[LL], *wkT[LL], *wvT[LL], *woT[LL], *w1T[LL], *w2T[LL];
    for (int l = 0; l < LL; ++l) {
        wqT[l] = hp; hp += (size_t)EE * EE;
        wkT[l] = hp; hp += (size_t)EE * EE;
        wvT[l] = hp; hp += (size_t)EE * EE;
        woT[l] = hp; hp += (size_t)EE * EE;
        w1T[l] = hp; hp += (size_t)EE * FFD;
        w2T[l] = hp; hp += (size_t)EE * FFD;
    }
    _Float16* wihT = hp; hp += (size_t)EE * 4 * EE;
    _Float16* whhT = hp; hp += (size_t)EE * 4 * EE;
    _Float16* dqT = hp; hp += (size_t)EE * EE;
    _Float16* dkT = hp; hp += (size_t)EE * EE;
    _Float16* dvT = hp; hp += (size_t)EE * EE;
    _Float16* doT = hp; hp += (size_t)EE * EE;
    _Float16* dpT = hp; hp += (size_t)EE * EE;
    int* visited  = (int*)hp;
    int* ucount   = visited + BB * NN;
    int* complete = ucount + BB;
    int* allvis   = complete + BB;
    int* firstn   = allvis + BB;
    int* prevn    = firstn + BB;

    // ---- one-time weight transforms (f32 -> transposed f16) ----
    for (int l = 0; l < LL; ++l) {
        xpose(stream, enc_wq + (size_t)l * EE * EE, wqT[l], EE, EE);
        xpose(stream, enc_wk + (size_t)l * EE * EE, wkT[l], EE, EE);
        xpose(stream, enc_wv + (size_t)l * EE * EE, wvT[l], EE, EE);
        xpose(stream, enc_wo + (size_t)l * EE * EE, woT[l], EE, EE);
        xpose(stream, enc_w1 + (size_t)l * EE * FFD, w1T[l], EE, FFD);
        xpose(stream, enc_w2 + (size_t)l * FFD * EE, w2T[l], FFD, EE);
    }
    xpose(stream, lstm_wih, wihT, EE, 4 * EE);
    xpose(stream, lstm_whh, whhT, EE, 4 * EE);
    xpose(stream, dec_wq, dqT, EE, EE);
    xpose(stream, dec_wk, dkT, EE, EE);
    xpose(stream, dec_wv, dvT, EE, EE);
    xpose(stream, dec_wo, doT, EE, EE);
    xpose(stream, dec_pw, dpT, EE, EE);

    // ---- input projection + positional encoding ----
    input_proj_k<<<(ROWS * EE) / 256, 256, 0, stream>>>(node_features, in_w, in_b, pe, xbuf, xh);

    // ---- encoder: 6 transformer layers ----
    for (int l = 0; l < LL; ++l) {
        gemm(stream, xh, wqT[l], enc_bq + (size_t)l * EE, nullptr, qb, nullptr, ROWS, EE, EE, 0, 0, 0);
        gemm(stream, xh, wkT[l], enc_bk + (size_t)l * EE, nullptr, kb, nullptr, ROWS, EE, EE, 0, 0, 0);
        gemm(stream, xh, wvT[l], enc_bv + (size_t)l * EE, nullptr, vb, nullptr, ROWS, EE, EE, 0, 0, 0);
        enc_attn_k<<<BB * HH, 128, 0, stream>>>(qb, kb, vb, abh);
        gemm(stream, abh, woT[l], enc_bo + (size_t)l * EE, xbuf, tb, nullptr, ROWS, EE, EE, 0, /*res*/1, 0);
        layernorm_k<<<ROWS / 8, 256, 0, stream>>>(tb, ln1g + (size_t)l * EE, ln1b + (size_t)l * EE, xbuf, xh, ROWS);
        gemm(stream, xh, w1T[l], enc_b1 + (size_t)l * FFD, nullptr, nullptr, ffh, ROWS, FFD, EE, /*relu*/1, 0, 0);
        gemm(stream, ffh, w2T[l], enc_b2 + (size_t)l * EE, xbuf, tb, nullptr, ROWS, EE, FFD, 0, /*res*/1, 0);
        layernorm_k<<<ROWS / 8, 256, 0, stream>>>(tb, ln2g + (size_t)l * EE, ln2b + (size_t)l * EE, xbuf, xh, ROWS);
    }
    // xbuf/xh == node_emb from here on

    // ---- decoder precompute ----
    gemm(stream, xh, dkT, dec_bk, nullptr, kb, nullptr, ROWS, EE, EE, 0, 0, 0);
    gemm(stream, xh, dvT, dec_bv, nullptr, vb, nullptr, ROWS, EE, EE, 0, 0, 0);
    ctx_init_k<<<(BB * EE + 255) / 256, 256, 0, stream>>>(xbuf, ctxh);
    zero_h_k<<<(BB * EE + 255) / 256, 256, 0, stream>>>(hh, BB * EE);
    zero_f_k<<<(BB * EE + 255) / 256, 256, 0, stream>>>(cb, BB * EE);
    zero_i_k<<<(BB * NN + 255) / 256, 256, 0, stream>>>(visited, BB * NN);
    zero_i_k<<<1, 256, 0, stream>>>(ucount, BB);
    zero_i_k<<<1, 256, 0, stream>>>(complete, BB);
    zero_i_k<<<1, 256, 0, stream>>>(allvis, BB);
    zero_i_k<<<1, 256, 0, stream>>>(firstn, BB);
    zero_i_k<<<1, 256, 0, stream>>>(prevn, BB);

    // ---- autoregressive decode ----
    for (int s = 0; s < MAX_STEPS; ++s) {
        gemm(stream, ctxh, wihT, lstm_bih, nullptr, gatesb, nullptr, BB, 4 * EE, EE, 0, 0, 0);
        gemm(stream, hh,   whhT, lstm_bhh, nullptr, gatesb, nullptr, BB, 4 * EE, EE, 0, 0, /*accum*/1);
        lstm_elem_k<<<(BB * EE + 255) / 256, 256, 0, stream>>>(gatesb, hb, cb, hh);
        done_update_k<<<1, 128, 0, stream>>>(hb, done_w, done_b, ucount, complete, s);
        gemm(stream, hh, dqT, dec_bq, nullptr, qdec, nullptr, BB, EE, EE, 0, 0, 0);
        dec_attn_k<<<BB * HH, 128, 0, stream>>>(qdec, kb, vb, adh);
        gemm(stream, adh, doT, dec_bo, nullptr, odec, nullptr, BB, EE, EE, 0, 0, 0);
        layernorm_k<<<BB / 8, 256, 0, stream>>>(odec, dec_lng, dec_lnb, nullptr, o2h, BB);
        gemm(stream, o2h, dpT, dec_pb, nullptr, lgt, nullptr, BB, EE, EE, 0, 0, 0);
        select_k<<<BB, 128, 0, stream>>>(lgt, xbuf, edge_weights,
                                         visited, ucount, complete, allvis, firstn, prevn,
                                         ctxh, shortcut_bonus, revisit_penalty,
                                         logps_f, tours_i, s);
    }
    finalize_k<<<1, 128, 0, stream>>>(complete, firstn, logps_f, tours_i);
}
